// PointAttentionBlock_34892314313405
// MI455X (gfx1250) — compile-verified, matrix operand order checked by reading, not run on hardware
//
#include <hip/hip_runtime.h>
#include <hip/hip_bf16.h>
#include <math.h>

// ---------------------------------------------------------------------------
// Types for CDNA5 WMMA (wave32): v16bf = 16 x bf16 (8 VGPRs), v8f = 8 x f32.
// ---------------------------------------------------------------------------
typedef __bf16 bf16_t;
typedef __bf16 v16bf __attribute__((ext_vector_type(16)));
typedef float  v8f   __attribute__((ext_vector_type(8)));

#define DEV __device__ __forceinline__

DEV float gelu_exact(float x) {
    return 0.5f * x * (1.0f + erff(x * 0.70710678118654752f));
}

// Load a 16x32 bf16 A/B fragment row for this lane from an LDS row.
// Per ISA 7.12.2 (16-bit A-matrix 16x32): lane L holds m = L%16 and, with
// half = L/16, K-values {k0+8h .. k0+8h+7} then {k0+16+8h .. k0+16+8h+7}
// packed pairwise into 8 VGPRs. Two 16B LDS reads.
DEV v16bf load_frag(const bf16_t* __restrict__ row, int k0, int half) {
    v16bf f;
    ((uint4*)&f)[0] = *(const uint4*)(row + k0 + 8 * half);
    ((uint4*)&f)[1] = *(const uint4*)(row + k0 + 16 + 8 * half);
    return f;
}

DEV v8f wmma_bf16(v16bf a, v16bf b, v8f c) {
    // D = A(16x32) x B(32x16) + C, f32 accumulate.
    return __builtin_amdgcn_wmma_f32_16x16x32_bf16(
        /*neg_a=*/false, a, /*neg_b=*/false, b,
        /*c_mod=*/(short)0, c, /*reuse_a=*/false, /*reuse_b=*/false);
}

// ---------------------------------------------------------------------------
// CDNA5 async global->LDS copy (GLOBAL_LOAD_ASYNC_TO_LDS_B128, ASYNCcnt).
// VDST = per-lane LDS byte address (low 32 bits of the flat shared pointer
// are the LDS offset per the aperture decode), VADDR = 64-bit global addr.
// ---------------------------------------------------------------------------
DEV void async_cp_b128(void* lds_ptr, const void* gptr) {
    unsigned int l = (unsigned int)(size_t)lds_ptr;
    asm volatile("global_load_async_to_lds_b128 %0, %1, off"
                 :: "v"(l), "v"(gptr)
                 : "memory");
}
DEV void wait_async0() {
    asm volatile("s_wait_asynccnt 0x0" ::: "memory");
}

// ---------------------------------------------------------------------------
// LayerNorm: one wave32 per 128-wide row. 8 rows per 256-thread block.
// ---------------------------------------------------------------------------
__global__ void __launch_bounds__(256)
k_layernorm(const float* __restrict__ in, const float* __restrict__ w,
            const float* __restrict__ b, bf16_t* __restrict__ out, int rows) {
    int wave = threadIdx.x >> 5, lane = threadIdx.x & 31;
    int row = blockIdx.x * 8 + wave;
    if (row >= rows) return;
    const float* p = in + (size_t)row * 128;
    float v0 = p[lane], v1 = p[lane + 32], v2 = p[lane + 64], v3 = p[lane + 96];
    float s = v0 + v1 + v2 + v3;
    for (int off = 16; off > 0; off >>= 1) s += __shfl_xor(s, off, 32);
    float mean = s * (1.0f / 128.0f);
    float d0 = v0 - mean, d1 = v1 - mean, d2 = v2 - mean, d3 = v3 - mean;
    float q = d0 * d0 + d1 * d1 + d2 * d2 + d3 * d3;
    for (int off = 16; off > 0; off >>= 1) q += __shfl_xor(q, off, 32);
    float rstd = rsqrtf(q * (1.0f / 128.0f) + 1e-5f);
    bf16_t* o = out + (size_t)row * 128;
    o[lane]      = (bf16_t)(d0 * rstd * w[lane]      + b[lane]);
    o[lane + 32] = (bf16_t)(d1 * rstd * w[lane + 32] + b[lane + 32]);
    o[lane + 64] = (bf16_t)(d2 * rstd * w[lane + 64] + b[lane + 64]);
    o[lane + 96] = (bf16_t)(d3 * rstd * w[lane + 96] + b[lane + 96]);
}

// ---------------------------------------------------------------------------
// Generic WMMA GEMM: C[M,N] = A[M,K](bf16) * W[K,N](f32 weights) + bias.
// Block tile 128x64, 8 waves, each wave owns a 32x32 tile (2x2 WMMA tiles).
// EPI: 0 = store bf16, 1 = gelu->bf16, 2 = +residual(f32) -> f32.
// ---------------------------------------------------------------------------
template <int EPI>
__global__ void __launch_bounds__(256)
k_gemm(const bf16_t* __restrict__ A, int lda,
       const float* __restrict__ W, int ldw,
       const float* __restrict__ bias,
       const float* __restrict__ res, int ldr,
       void* __restrict__ outv, int ldo, int K) {
    __shared__ bf16_t As[128][32];   // A tile, row-major
    __shared__ bf16_t Bts[64][32];   // W tile transposed: Bts[n][k] = W[k][n]

    int tid = threadIdx.x;
    int m0 = blockIdx.x * 128, n0 = blockIdx.y * 64;
    int wave = tid >> 5, lane = tid & 31, half = lane >> 4, ln = lane & 15;
    int wm = wave & 3, wn = wave >> 2;  // 4x2 wave grid over 128x64

    v8f acc[2][2] = {};

    for (int k0 = 0; k0 < K; k0 += 32) {
        // Stage A tile: each thread copies 16 contiguous bf16 (two uint4).
        {
            int r = tid >> 1, c = (tid & 1) * 16;
            const uint4* src = (const uint4*)(A + (size_t)(m0 + r) * lda + k0 + c);
            uint4* dst = (uint4*)&As[r][c];
            dst[0] = src[0];
            dst[1] = src[1];
        }
        // Stage W tile transposed + f32->bf16 convert (weights are L2-hot).
        {
            int n = tid >> 2, ks = (tid & 3) * 8;
            for (int i = 0; i < 8; ++i)
                Bts[n][ks + i] = (bf16_t)W[(size_t)(k0 + ks + i) * ldw + n0 + n];
        }
        __syncthreads();

        v16bf a0 = load_frag(&As[wm * 32 + ln][0], 0, half);
        v16bf a1 = load_frag(&As[wm * 32 + 16 + ln][0], 0, half);
        v16bf b0 = load_frag(&Bts[wn * 32 + ln][0], 0, half);
        v16bf b1 = load_frag(&Bts[wn * 32 + 16 + ln][0], 0, half);
        acc[0][0] = wmma_bf16(a0, b0, acc[0][0]);
        acc[0][1] = wmma_bf16(a0, b1, acc[0][1]);
        acc[1][0] = wmma_bf16(a1, b0, acc[1][0]);
        acc[1][1] = wmma_bf16(a1, b1, acc[1][1]);
        __syncthreads();
    }

    // Epilogue. D element r of lane: m = r + 8*half, n = lane%16 (ISA 7.12.2).
    for (int i = 0; i < 2; ++i)
        for (int j = 0; j < 2; ++j)
            for (int r = 0; r < 8; ++r) {
                int grow = m0 + wm * 32 + i * 16 + r + 8 * half;
                int gcol = n0 + wn * 32 + j * 16 + ln;
                float v = acc[i][j][r] + bias[gcol];
                if (EPI == 0) {
                    ((bf16_t*)outv)[(size_t)grow * ldo + gcol] = (bf16_t)v;
                } else if (EPI == 1) {
                    ((bf16_t*)outv)[(size_t)grow * ldo + gcol] = (bf16_t)gelu_exact(v);
                } else {
                    ((float*)outv)[(size_t)grow * ldo + gcol] =
                        v + res[(size_t)grow * ldr + gcol];
                }
            }
}

// ---------------------------------------------------------------------------
// Fused attention with per-pair positional bias (flash-style online softmax).
// Grid: (96, 4) = (B * N/16 query tiles, 4 key segments of 12 tiles each).
// 256 threads = 8 waves. pb tile (the dominant 38.7 GFLOP GEMM) is computed
// once in LDS and reused by all 4 heads; it never touches HBM (saves ~0.9 GB).
// k/v tiles are double-buffered and prefetched with async global->LDS copies
// (ASYNCcnt), overlapping the next tile's 8 KB staging with h1+pb compute.
// ---------------------------------------------------------------------------
__global__ void __launch_bounds__(256)
k_attn(const bf16_t* __restrict__ Qb, const bf16_t* __restrict__ Kb,
       const bf16_t* __restrict__ Vb, const float* __restrict__ coords,
       const float* __restrict__ Wp1, const float* __restrict__ bp1,
       const float* __restrict__ Wp2, const float* __restrict__ bp2,
       float* __restrict__ part_acc, float* __restrict__ part_m,
       float* __restrict__ part_l) {
    __shared__ float  Wp1s[3][128];
    __shared__ float  bp1s[128];
    __shared__ float  bp2s[128];
    __shared__ bf16_t Wp2t[128][128];    // Wp2 transposed, bf16 (B operand)
    __shared__ bf16_t qs[16][128];
    __shared__ bf16_t ks[2][16][128];    // double-buffered (async prefetch)
    __shared__ bf16_t vs[2][16][128];
    __shared__ bf16_t h1[256][128];      // gelu(rel@Wp1+bp1), 256 pairs
    __shared__ bf16_t pb[256][128];      // h1@Wp2+bp2
    __shared__ float  logit[4][16][16];
    __shared__ float  probs[4][16][16];
    __shared__ float  alph[4][16];
    __shared__ float  mrow[4][16];
    __shared__ float  lrow[4][16];
    __shared__ float  accs[4][16][32];

    int tid = threadIdx.x;
    int wg = blockIdx.x;   // 0..95 : b*48 + query tile
    int seg = blockIdx.y;  // 0..3
    int b = wg / 48, qt = wg % 48;
    int qrow0 = b * 768 + qt * 16;

    // Per-thread copy slot for k/v staging: 16 B (8 bf16) per thread.
    int ce = tid * 8, cr = ce >> 7, cc = ce & 127;

    // ---- kick off async prefetch of tile 0 before the one-time staging ----
    {
        int krow0 = b * 768 + (seg * 12) * 16;
        async_cp_b128(&ks[0][cr][cc], Kb + (size_t)(krow0 + cr) * 128 + cc);
        async_cp_b128(&vs[0][cr][cc], Vb + (size_t)(krow0 + cr) * 128 + cc);
    }

    // ---- one-time staging ----
    for (int i = tid; i < 384; i += 256) Wp1s[i / 128][i % 128] = Wp1[i];
    for (int i = tid; i < 128; i += 256) { bp1s[i] = bp1[i]; bp2s[i] = bp2[i]; }
    for (int i = tid; i < 16384; i += 256) {
        int k = i >> 7, n = i & 127;
        Wp2t[n][k] = (bf16_t)Wp2[i];
    }
    *(uint4*)&qs[cr][cc] = *(const uint4*)(Qb + (size_t)(qrow0 + cr) * 128 + cc);
    if (tid < 64) { mrow[tid >> 4][tid & 15] = -3.0e38f; lrow[tid >> 4][tid & 15] = 0.f; }
    for (int i = tid; i < 2048; i += 256) ((float*)accs)[i] = 0.f;

    // Per-thread query coordinates (pair mapping: qi = tid/16, kj = tid%16).
    int qi_p = tid >> 4, kj_p = tid & 15;
    float qc0 = coords[(size_t)(qrow0 + qi_p) * 3 + 0];
    float qc1 = coords[(size_t)(qrow0 + qi_p) * 3 + 1];
    float qc2 = coords[(size_t)(qrow0 + qi_p) * 3 + 2];

    for (int t = 0; t < 12; ++t) {
        int cur = t & 1;
        int krow0 = b * 768 + (seg * 12 + t) * 16;

        // Tile t data ready: retire this wave's async copies, then barrier so
        // every wave sees the LDS writes (and all reads of buf cur^1 from the
        // previous iteration are ordered before the prefetch below).
        wait_async0();
        __syncthreads();

        if (t < 11) {   // prefetch tile t+1 into the other buffer
            int nrow0 = b * 768 + (seg * 12 + t + 1) * 16;
            async_cp_b128(&ks[cur ^ 1][cr][cc], Kb + (size_t)(nrow0 + cr) * 128 + cc);
            async_cp_b128(&vs[cur ^ 1][cr][cc], Vb + (size_t)(nrow0 + cr) * 128 + cc);
        }

        // ---- h1 = gelu(rel @ Wp1 + bp1), one pair per thread (K=3: VALU) ----
        {
            const float* kc = coords + (size_t)(krow0 + kj_p) * 3;
            float r0 = qc0 - kc[0], r1 = qc1 - kc[1], r2 = qc2 - kc[2];
            for (int i = 0; i < 128; ++i) {
                int c = (i + (tid & 31) * 4) & 127;  // stagger -> bank-spread
                float v = bp1s[c] + r0 * Wp1s[0][c] + r1 * Wp1s[1][c] + r2 * Wp1s[2][c];
                h1[tid][c] = (bf16_t)gelu_exact(v);
            }
        }
        __syncthreads();

        // ---- pb = h1[256,128] @ Wp2[128,128] + bp2 : the dominant GEMM ----
        // Hoist A fragments for both m-subtiles; share each B fragment
        // across two WMMAs (halves LDS B traffic).
        {
            int wave = tid >> 5, lane = tid & 31, half = lane >> 4, ln = lane & 15;
            const bf16_t* arow0 = &h1[(wave * 2 + 0) * 16 + ln][0];
            const bf16_t* arow1 = &h1[(wave * 2 + 1) * 16 + ln][0];
            v16bf a0[4], a1[4];
            for (int kk = 0; kk < 4; ++kk) {
                a0[kk] = load_frag(arow0, kk * 32, half);
                a1[kk] = load_frag(arow1, kk * 32, half);
            }
            for (int nt = 0; nt < 8; ++nt) {
                v8f c0 = {}, c1 = {};
                const bf16_t* brow = &Wp2t[nt * 16 + ln][0];
                for (int kk = 0; kk < 4; ++kk) {
                    v16bf bfr = load_frag(brow, kk * 32, half);
                    c0 = wmma_bf16(a0[kk], bfr, c0);
                    c1 = wmma_bf16(a1[kk], bfr, c1);
                }
                int n = nt * 16 + ln;
                float bb = bp2s[n];
                for (int r = 0; r < 8; ++r) {
                    pb[(wave * 2 + 0) * 16 + r + 8 * half][n] = (bf16_t)(c0[r] + bb);
                    pb[(wave * 2 + 1) * 16 + r + 8 * half][n] = (bf16_t)(c1[r] + bb);
                }
            }
        }
        __syncthreads();

        // ---- logits = SCALE * q·(k + pb)  (pb term is diagonal-in-n: VALU) ----
        {
            for (int h = 0; h < 4; ++h) {
                int c0 = h * 32;
                float s = 0.f;
                for (int d = 0; d < 32; ++d) {
                    float qv = (float)qs[qi_p][c0 + d];
                    s += qv * ((float)ks[cur][kj_p][c0 + d] + (float)pb[tid][c0 + d]);
                }
                logit[h][qi_p][kj_p] = s * 0.17677669529663687f;  // 1/sqrt(32)
            }
        }
        __syncthreads();

        // ---- online softmax state update (64 rows = 4 heads x 16 queries) ----
        if (tid < 64) {
            int h = tid >> 4, qi = tid & 15;
            float mx = logit[h][qi][0];
            for (int kj = 1; kj < 16; ++kj) mx = fmaxf(mx, logit[h][qi][kj]);
            float mold = mrow[h][qi];
            float mnew = fmaxf(mold, mx);
            float a = __expf(mold - mnew);
            float ps = 0.f;
            for (int kj = 0; kj < 16; ++kj) {
                float pv = __expf(logit[h][qi][kj] - mnew);
                probs[h][qi][kj] = pv;
                ps += pv;
            }
            lrow[h][qi] = lrow[h][qi] * a + ps;
            alph[h][qi] = a;
            mrow[h][qi] = mnew;
        }
        __syncthreads();

        // ---- acc = acc*alpha + probs·(v + pb)  (diagonal-in-n: VALU) ----
        {
            int h = tid >> 6, rem = tid & 63, qi = rem >> 2, dg = (rem & 3) * 8;
            float a = alph[h][qi];
            for (int dd = 0; dd < 8; ++dd) {
                int d = dg + dd, c = h * 32 + d;
                float acc0 = accs[h][qi][d] * a;
                for (int kj = 0; kj < 16; ++kj)
                    acc0 += probs[h][qi][kj] *
                            ((float)vs[cur][kj][c] + (float)pb[qi * 16 + kj][c]);
                accs[h][qi][d] = acc0;
            }
        }
        // no trailing barrier: next iteration's tile-ready barrier orders
        // these reads before the buffer is overwritten.
    }

    // ---- write flash partials for this segment (same-thread data only) ----
    size_t base = ((size_t)(wg * 4 + seg)) * 2048;
    {
        int h = tid >> 6, rem = tid & 63, qi = rem >> 2, dg = (rem & 3) * 8;
        for (int dd = 0; dd < 8; ++dd) {
            int d = dg + dd;
            part_acc[base + (size_t)(h * 16 + qi) * 32 + d] = accs[h][qi][d];
        }
    }
    if (tid < 64) {
        int h = tid >> 4, qi = tid & 15;
        size_t bm = ((size_t)(wg * 4 + seg)) * 64 + h * 16 + qi;
        part_m[bm] = mrow[h][qi];
        part_l[bm] = lrow[h][qi];
    }
}

// ---------------------------------------------------------------------------
// Merge the 4 flash segments and emit attended (heads merged) as bf16 [BN,H].
// ---------------------------------------------------------------------------
__global__ void __launch_bounds__(64)
k_merge(const float* __restrict__ part_acc, const float* __restrict__ part_m,
        const float* __restrict__ part_l, bf16_t* __restrict__ attended) {
    int wg = blockIdx.x, tid = threadIdx.x;
    int h = tid >> 4, qi = tid & 15;
    int b = wg / 48, qt = wg % 48;
    float ms[4], ls[4], M = -3.0e38f;
    for (int s = 0; s < 4; ++s) {
        size_t bm = ((size_t)(wg * 4 + s)) * 64 + h * 16 + qi;
        ms[s] = part_m[bm];
        ls[s] = part_l[bm];
        M = fmaxf(M, ms[s]);
    }
    float L = 0.f, wsc[4];
    for (int s = 0; s < 4; ++s) { wsc[s] = __expf(ms[s] - M); L += ls[s] * wsc[s]; }
    float invL = 1.0f / L;
    int row = b * 768 + qt * 16 + qi;
    for (int d = 0; d < 32; ++d) {
        float v = 0.f;
        for (int s = 0; s < 4; ++s) {
            size_t ba = ((size_t)(wg * 4 + s)) * 2048 + (size_t)(h * 16 + qi) * 32 + d;
            v += wsc[s] * part_acc[ba];
        }
        attended[(size_t)row * 128 + h * 32 + d] = (bf16_t)(v * invL);
    }
}

// ---------------------------------------------------------------------------
// Host-side orchestration (all on `stream`, graph-capture safe).
// ---------------------------------------------------------------------------
extern "C" void kernel_launch(void* const* d_in, const int* in_sizes, int n_in,
                              void* d_out, int out_size, void* d_ws, size_t ws_size,
                              hipStream_t stream) {
    (void)in_sizes; (void)n_in; (void)out_size; (void)ws_size;
    const float* features = (const float*)d_in[0];
    const float* coords   = (const float*)d_in[1];
    const float* ln1_w = (const float*)d_in[2];
    const float* ln1_b = (const float*)d_in[3];
    const float* Wq = (const float*)d_in[4];  const float* bq = (const float*)d_in[5];
    const float* Wk = (const float*)d_in[6];  const float* bk = (const float*)d_in[7];
    const float* Wv = (const float*)d_in[8];  const float* bv = (const float*)d_in[9];
    const float* Wp1 = (const float*)d_in[10]; const float* bp1 = (const float*)d_in[11];
    const float* Wp2 = (const float*)d_in[12]; const float* bp2 = (const float*)d_in[13];
    const float* Wo = (const float*)d_in[14]; const float* bo = (const float*)d_in[15];
    const float* ln2_w = (const float*)d_in[16];
    const float* ln2_b = (const float*)d_in[17];
    const float* Wf1 = (const float*)d_in[18]; const float* bf1 = (const float*)d_in[19];
    const float* Wf2 = (const float*)d_in[20]; const float* bf2 = (const float*)d_in[21];

    // Workspace carve-out (~7.3 MB, 256B-aligned slabs).
    char* wsp = (char*)d_ws;
    size_t off = 0;
    auto alloc = [&](size_t bytes) -> void* {
        void* p = wsp + off;
        off += (bytes + 255) & ~((size_t)255);
        return p;
    };
    const int BN = 1536;  // B*N
    bf16_t* xhat  = (bf16_t*)alloc((size_t)BN * 128 * 2);
    bf16_t* Qb    = (bf16_t*)alloc((size_t)BN * 128 * 2);
    bf16_t* Kb    = (bf16_t*)alloc((size_t)BN * 128 * 2);
    bf16_t* Vb    = (bf16_t*)alloc((size_t)BN * 128 * 2);
    bf16_t* att   = (bf16_t*)alloc((size_t)BN * 128 * 2);
    float*  feat2 = (float*)alloc((size_t)BN * 128 * 4);
    bf16_t* y2    = (bf16_t*)alloc((size_t)BN * 128 * 2);
    bf16_t* ffh   = (bf16_t*)alloc((size_t)BN * 256 * 2);
    float*  p_acc = (float*)alloc((size_t)96 * 4 * 2048 * 4);
    float*  p_m   = (float*)alloc((size_t)96 * 4 * 64 * 4);
    float*  p_l   = (float*)alloc((size_t)96 * 4 * 64 * 4);

    dim3 blk(256);
    dim3 g128(12, 2);  // M/128=12, N/64=2 (N=128)
    dim3 g256(12, 4);  // N=256
    dim3 gatt(96, 4);

    // x = LN1(features)
    k_layernorm<<<192, blk, 0, stream>>>(features, ln1_w, ln1_b, xhat, BN);
    // q,k,v projections (WMMA)
    k_gemm<0><<<g128, blk, 0, stream>>>(xhat, 128, Wq, 128, bq, nullptr, 0, Qb, 128, 128);
    k_gemm<0><<<g128, blk, 0, stream>>>(xhat, 128, Wk, 128, bk, nullptr, 0, Kb, 128, 128);
    k_gemm<0><<<g128, blk, 0, stream>>>(xhat, 128, Wv, 128, bv, nullptr, 0, Vb, 128, 128);
    // fused pair-bias attention (flash partials) + merge
    k_attn<<<gatt, blk, 0, stream>>>(Qb, Kb, Vb, coords, Wp1, bp1, Wp2, bp2,
                                     p_acc, p_m, p_l);
    k_merge<<<96, 64, 0, stream>>>(p_acc, p_m, p_l, att);
    // feat2 = features + attended @ Wo + bo
    k_gemm<2><<<g128, blk, 0, stream>>>(att, 128, Wo, 128, bo, features, 128,
                                        feat2, 128, 128);
    // y = LN2(feat2); ffh = gelu(y @ Wf1 + bf1); out = feat2 + ffh @ Wf2 + bf2
    k_layernorm<<<192, blk, 0, stream>>>(feat2, ln2_w, ln2_b, y2, BN);
    k_gemm<1><<<g256, blk, 0, stream>>>(y2, 128, Wf1, 256, bf1, nullptr, 0,
                                        ffh, 256, 128);
    k_gemm<2><<<g128, blk, 0, stream>>>(ffh, 256, Wf2, 128, bf2, feat2, 128,
                                        d_out, 128, 256);
}
